// MoeLayer_60842506715596
// MI455X (gfx1250) — matrix operand order repaired and checked by reading, not compiled, not run on hardware
//
#include <hip/hip_runtime.h>
#include <hip/hip_bf16.h>

#define TOK   8192
#define DIMC  1024
#define HIDC  2048
#define NEXP  8

typedef __attribute__((ext_vector_type(16))) __bf16 v16bf;
typedef __attribute__((ext_vector_type(8)))  float  v8f;

union Frag {
    uint4 q[2];
    v16bf v;
};

__device__ inline unsigned short f2bf(float f) {
    unsigned int u = __float_as_uint(f);
    u += 0x7FFFu + ((u >> 16) & 1u);      // round-to-nearest-even
    return (unsigned short)(u >> 16);
}

__device__ inline unsigned pack2bf(float lo, float hi) {
    union { __hip_bfloat162 b; unsigned u; } cvt;
    cvt.b = __float22bfloat162_rn(float2{lo, hi});   // v_cvt_pk_bf16_f32
    return cvt.u;
}

__device__ inline float fast_sigmoid(float z) {
    return __builtin_amdgcn_rcpf(1.f + __expf(-z));  // v_rcp_f32, no IEEE div
}

__device__ inline void atomAddF(float* p, float v) {
    unsafeAtomicAdd(p, v);                // global_atomic_add_f32
}

// ---------------------------------------------------------------------------
// CDNA5 async global->LDS copy (GLOBAL_LOAD_ASYNC_TO_LDS_B128, ASYNCcnt).
// lds_addr: LDS byte address (low 32 bits of a generic __shared__ pointer).
// ---------------------------------------------------------------------------
__device__ __forceinline__ void async_b128(unsigned lds_addr, const void* g) {
    asm volatile("global_load_async_to_lds_b128 %0, %1, off"
                 :: "v"(lds_addr), "v"(g) : "memory");
}
__device__ __forceinline__ void wait_async0() {
#if __has_builtin(__builtin_amdgcn_s_wait_asynccnt)
    __builtin_amdgcn_s_wait_asynccnt(0);
#else
    asm volatile("s_wait_asynccnt 0x0" ::: "memory");
#endif
}
__device__ __forceinline__ unsigned lds_addr_of(const void* p) {
    return (unsigned)(uintptr_t)p;        // LDS aperture: low 32 bits = LDS offset
}

// ---------------------------------------------------------------------------
// 0a) x fp32 -> bf16 (row-major unchanged)
// ---------------------------------------------------------------------------
__global__ __launch_bounds__(256) void xconv_kernel(
    const float* __restrict__ x, unsigned short* __restrict__ xb)
{
    size_t t = (size_t)blockIdx.x * 256 + threadIdx.x;
    float4 f0 = *(const float4*)(x + t * 8);
    float4 f1 = *(const float4*)(x + t * 8 + 4);
    uint4 q;
    q.x = pack2bf(f0.x, f0.y);
    q.y = pack2bf(f0.z, f0.w);
    q.z = pack2bf(f1.x, f1.y);
    q.w = pack2bf(f1.z, f1.w);
    ((uint4*)xb)[t] = q;
}

// ---------------------------------------------------------------------------
// 0b) Weight fp32 [E][K][N] -> bf16 transposed [E][N][K].
// ---------------------------------------------------------------------------
template <int K, int N>
__global__ __launch_bounds__(256) void wconv_kernel(
    const float* __restrict__ in, unsigned short* __restrict__ outT)
{
    size_t tid = (size_t)blockIdx.x * 256 + threadIdx.x;
    int n  = (int)(tid & (N - 1));
    size_t q = tid / N;
    int kb = (int)(q & (K / 8 - 1));      // 8 k's per thread
    int e  = (int)(q / (K / 8));
    const float* src = in + ((size_t)e * K + kb * 8) * N + n;
    uint4 o;
    o.x = pack2bf(src[0 * (size_t)N], src[1 * (size_t)N]);
    o.y = pack2bf(src[2 * (size_t)N], src[3 * (size_t)N]);
    o.z = pack2bf(src[4 * (size_t)N], src[5 * (size_t)N]);
    o.w = pack2bf(src[6 * (size_t)N], src[7 * (size_t)N]);
    *(uint4*)&outT[((size_t)e * N + n) * K + kb * 8] = o;
}

// ---------------------------------------------------------------------------
// 1) Gating: one wave per token. logits = x @ gate_w.T, top-2 softmax.
// ---------------------------------------------------------------------------
__global__ __launch_bounds__(256) void gate_kernel(
    const float* __restrict__ x, const float* __restrict__ gw,
    int* __restrict__ tok_e, float* __restrict__ tok_w, int* __restrict__ counts)
{
    __shared__ float sg[NEXP * DIMC];     // 32 KB
    for (int i = threadIdx.x; i < NEXP * DIMC; i += 256) sg[i] = gw[i];
    __syncthreads();

    const int wave = threadIdx.x >> 5;
    const int lane = threadIdx.x & 31;
    const int t    = blockIdx.x * 8 + wave;

    float acc[NEXP];
#pragma unroll
    for (int e = 0; e < NEXP; ++e) acc[e] = 0.f;

    for (int d = lane; d < DIMC; d += 32) {
        float xv = x[(size_t)t * DIMC + d];
#pragma unroll
        for (int e = 0; e < NEXP; ++e) acc[e] += xv * sg[e * DIMC + d];
    }
#pragma unroll
    for (int off = 16; off > 0; off >>= 1) {
#pragma unroll
        for (int e = 0; e < NEXP; ++e) acc[e] += __shfl_xor(acc[e], off, 32);
    }
    if (lane == 0) {
        int e0 = 0;
        float v0 = acc[0];
#pragma unroll
        for (int e = 1; e < NEXP; ++e) if (acc[e] > v0) { v0 = acc[e]; e0 = e; }
        int e1 = -1;
        float v1 = -3.4e38f;
#pragma unroll
        for (int e = 0; e < NEXP; ++e) if (e != e0 && acc[e] > v1) { v1 = acc[e]; e1 = e; }
        float w0 = fast_sigmoid(v0 - v1);
        float w1 = 1.f - w0;
        tok_e[t * 2 + 0] = e0; tok_e[t * 2 + 1] = e1;
        tok_w[t * 2 + 0] = w0; tok_w[t * 2 + 1] = w1;
        atomicAdd(&counts[e0], 1);
        atomicAdd(&counts[e1], 1);
    }
}

__global__ void scan_kernel(const int* __restrict__ counts, int* __restrict__ offsets)
{
    if (threadIdx.x == 0) {
        int s = 0;
        for (int e = 0; e < NEXP; ++e) { offsets[e] = s; s += counts[e]; }
    }
}

__global__ __launch_bounds__(256) void scatter_kernel(
    const int* __restrict__ tok_e, const float* __restrict__ tok_w,
    const int* __restrict__ offsets, int* __restrict__ cursors,
    int* __restrict__ row_token, float* __restrict__ row_weight)
{
    const int t = blockIdx.x * 256 + threadIdx.x;
#pragma unroll
    for (int k = 0; k < 2; ++k) {
        int   e = tok_e[t * 2 + k];
        float w = tok_w[t * 2 + k];
        int slot = offsets[e] + atomicAdd(&cursors[e], 1);
        row_token[slot]  = t;
        row_weight[slot] = w;
    }
}

// ---------------------------------------------------------------------------
// WMMA fragment helpers (ISA 7.12.2 layouts).
// ---------------------------------------------------------------------------
__device__ inline Frag loadA(const unsigned short* s, int mrow, int lane) {
    Frag f;
    int klo = (lane & 16) ? 8 : 0;
    f.q[0] = *(const uint4*)&s[mrow * 32 + klo];
    f.q[1] = *(const uint4*)&s[mrow * 32 + klo + 16];
    return f;
}
__device__ inline Frag loadB(const unsigned short* s, int nrow, int lane) {
    Frag f;
    int kb = (lane & 16) ? 8 : 0;
    f.q[0] = *(const uint4*)&s[nrow * 32 + kb];
    f.q[1] = *(const uint4*)&s[nrow * 32 + kb + 16];
    return f;
}
#define WMMA_BF16(A, B, C) \
    __builtin_amdgcn_wmma_f32_16x16x32_bf16(false, (A), false, (B), (short)0, (C), false, false)

// ---------------------------------------------------------------------------
// 4) GEMM1: H[slot, n] = silu(x@w1) * (x@w3). bf16 inputs.
//    Double-buffered LDS filled by ASYNC global->LDS copies (no data VGPRs):
//    barrier -> async(next) -> WMMA(cur) -> s_wait_asynccnt 0.
// ---------------------------------------------------------------------------
__global__ __launch_bounds__(256, 1) void gemm1_kernel(
    const unsigned short* __restrict__ xb,  const unsigned short* __restrict__ w1t,
    const unsigned short* __restrict__ w3t, const int* __restrict__ counts,
    const int* __restrict__ offsets, const int* __restrict__ row_token,
    unsigned short* __restrict__ Hbuf)
{
    const int e   = blockIdx.z;
    const int cnt = counts[e];
    const int m0  = blockIdx.y * 64;
    if (m0 >= cnt) return;
    const int off = offsets[e];
    const int n0  = blockIdx.x * 128;

    __shared__ unsigned short sX [2][64 * 32];    //  8 KB
    __shared__ unsigned short sW1[2][128 * 32];   // 16 KB, [n][k]
    __shared__ unsigned short sW3[2][128 * 32];   // 16 KB
    __shared__ int sTok[64];

    const int tid   = threadIdx.x;
    const int lane  = tid & 31;
    const int wave  = tid >> 5;
    const int waveM = wave >> 2;   // 0..1
    const int waveN = wave & 3;    // 0..3

    if (tid < 64) {
        int r = m0 + tid;
        if (r >= cnt) r = cnt - 1;
        sTok[tid] = row_token[off + r];
    }
    __syncthreads();

    // loop-invariant bases (all bf16, [n][k] for weights)
    const int rq = tid >> 2, cq = (tid & 3) * 8;
    const unsigned short* xbase = xb + (size_t)sTok[rq] * DIMC + cq;
    const unsigned short* w1b[2], * w3b[2];
#pragma unroll
    for (int i = 0; i < 2; ++i) {
        size_t nrow = (size_t)e * HIDC + n0 + rq + 64 * i;
        w1b[i] = w1t + nrow * DIMC + cq;
        w3b[i] = w3t + nrow * DIMC + cq;
    }
    // per-thread LDS byte addresses
    const unsigned lX  = lds_addr_of(sX)  + tid * 16u;
    const unsigned lW1 = lds_addr_of(sW1) + tid * 16u;
    const unsigned lW3 = lds_addr_of(sW3) + tid * 16u;

    auto stage = [&](int kk, int b) {   // async copies, no data registers
        async_b128(lX + b * 4096u, xbase + kk);
#pragma unroll
        for (int i = 0; i < 2; ++i) {
            async_b128(lW1 + b * 8192u + i * 4096u, w1b[i] + kk);
            async_b128(lW3 + b * 8192u + i * 4096u, w3b[i] + kk);
        }
    };

    v8f acc1[2][2], acc3[2][2];
#pragma unroll
    for (int a = 0; a < 2; ++a)
#pragma unroll
        for (int b = 0; b < 2; ++b) { acc1[a][b] = (v8f)0.f; acc3[a][b] = (v8f)0.f; }

    stage(0, 0);
    wait_async0();
    int buf = 0;
#pragma unroll 1
    for (int kk = 0; kk < DIMC; kk += 32, buf ^= 1) {
        __syncthreads();                    // buf data visible; prev reads of buf^1 done
        const bool more = (kk + 32 < DIMC);
        if (more) stage(kk + 32, buf ^ 1);  // async copies overlap WMMAs below

        Frag A[2];
#pragma unroll
        for (int mi = 0; mi < 2; ++mi)
            A[mi] = loadA(sX[buf], waveM * 32 + mi * 16 + (lane & 15), lane);
#pragma unroll
        for (int ni = 0; ni < 2; ++ni) {
            int nrow = waveN * 32 + ni * 16 + (lane & 15);
            Frag B1 = loadB(sW1[buf], nrow, lane);
            Frag B3 = loadB(sW3[buf], nrow, lane);
#pragma unroll
            for (int mi = 0; mi < 2; ++mi) {
                acc1[mi][ni] = WMMA_BF16(A[mi].v, B1.v, acc1[mi][ni]);
                acc3[mi][ni] = WMMA_BF16(A[mi].v, B3.v, acc3[mi][ni]);
            }
        }
        if (more) wait_async0();            // my async writes done before next barrier
    }

    // epilogue: silu(h) * g -> bf16 H
    const int mtop = (lane & 16) ? 8 : 0;
#pragma unroll
    for (int mi = 0; mi < 2; ++mi) {
#pragma unroll
        for (int ni = 0; ni < 2; ++ni) {
            int col = n0 + waveN * 32 + ni * 16 + (lane & 15);
#pragma unroll
            for (int v = 0; v < 8; ++v) {
                int row = m0 + waveM * 32 + mi * 16 + mtop + v;
                if (row < cnt) {
                    float h = acc1[mi][ni][v];
                    float g = acc3[mi][ni][v];
                    float val = h * fast_sigmoid(h) * g;
                    Hbuf[(size_t)(off + row) * HIDC + col] = f2bf(val);
                }
            }
        }
    }
}

// ---------------------------------------------------------------------------
// 5) GEMM2: out[tok] += weight * (H @ w2). bf16 inputs, K = 2048.
// ---------------------------------------------------------------------------
__global__ __launch_bounds__(256, 1) void gemm2_kernel(
    const unsigned short* __restrict__ Hbuf, const unsigned short* __restrict__ w2t,
    const int* __restrict__ counts, const int* __restrict__ offsets,
    const int* __restrict__ row_token, const float* __restrict__ row_weight,
    float* __restrict__ out)
{
    const int e   = blockIdx.z;
    const int cnt = counts[e];
    const int m0  = blockIdx.y * 64;
    if (m0 >= cnt) return;
    const int off = offsets[e];
    const int n0  = blockIdx.x * 128;

    __shared__ unsigned short sA[2][64 * 32];     //  8 KB
    __shared__ unsigned short sW[2][128 * 32];    // 16 KB, [n][k]
    __shared__ int   sTok[64];
    __shared__ float sWgt[64];

    const int tid   = threadIdx.x;
    const int lane  = tid & 31;
    const int wave  = tid >> 5;
    const int waveM = wave >> 2;
    const int waveN = wave & 3;

    if (tid < 64) {
        int r = m0 + tid;
        if (r >= cnt) r = cnt - 1;
        sTok[tid] = row_token[off + r];
        sWgt[tid] = row_weight[off + r];
    }
    __syncthreads();

    const int rq = tid >> 2, cq = (tid & 3) * 8;
    int rr = m0 + rq; if (rr >= cnt) rr = cnt - 1;
    const unsigned short* abase = Hbuf + (size_t)(off + rr) * HIDC + cq;
    const unsigned short* w2b[2];
#pragma unroll
    for (int i = 0; i < 2; ++i)
        w2b[i] = w2t + ((size_t)e * DIMC + n0 + rq + 64 * i) * HIDC + cq;

    const unsigned lA = lds_addr_of(sA) + tid * 16u;
    const unsigned lW = lds_addr_of(sW) + tid * 16u;

    auto stage = [&](int kk, int b) {
        async_b128(lA + b * 4096u, abase + kk);
#pragma unroll
        for (int i = 0; i < 2; ++i)
            async_b128(lW + b * 8192u + i * 4096u, w2b[i] + kk);
    };

    v8f acc[2][2];
#pragma unroll
    for (int a = 0; a < 2; ++a)
#pragma unroll
        for (int b = 0; b < 2; ++b) acc[a][b] = (v8f)0.f;

    stage(0, 0);
    wait_async0();
    int buf = 0;
#pragma unroll 1
    for (int kk = 0; kk < HIDC; kk += 32, buf ^= 1) {
        __syncthreads();
        const bool more = (kk + 32 < HIDC);
        if (more) stage(kk + 32, buf ^ 1);

        Frag A[2];
#pragma unroll
        for (int mi = 0; mi < 2; ++mi)
            A[mi] = loadA(sA[buf], waveM * 32 + mi * 16 + (lane & 15), lane);
#pragma unroll
        for (int ni = 0; ni < 2; ++ni) {
            Frag B = loadB(sW[buf], waveN * 32 + ni * 16 + (lane & 15), lane);
#pragma unroll
            for (int mi = 0; mi < 2; ++mi)
                acc[mi][ni] = WMMA_BF16(A[mi].v, B.v, acc[mi][ni]);
        }
        if (more) wait_async0();
    }

    const int mtop = (lane & 16) ? 8 : 0;
#pragma unroll
    for (int mi = 0; mi < 2; ++mi) {
#pragma unroll
        for (int ni = 0; ni < 2; ++ni) {
            int col = n0 + waveN * 32 + ni * 16 + (lane & 15);
#pragma unroll
            for (int v = 0; v < 8; ++v) {
                int rloc = waveM * 32 + mi * 16 + mtop + v;
                int row  = m0 + rloc;
                if (row < cnt) {
                    float val = acc[mi][ni][v] * sWgt[rloc];
                    atomAddF(&out[(size_t)sTok[rloc] * DIMC + col], val);
                }
            }
        }
    }
}

// ---------------------------------------------------------------------------
// Host launcher.  Workspace (~177 MB): see layout comments in previous rounds.
// ---------------------------------------------------------------------------
extern "C" void kernel_launch(void* const* d_in, const int* in_sizes, int n_in,
                              void* d_out, int out_size, void* d_ws, size_t ws_size,
                              hipStream_t stream) {
    const float* x  = (const float*)d_in[0];
    const float* gw = (const float*)d_in[1];
    const float* w1 = (const float*)d_in[2];
    const float* w3 = (const float*)d_in[3];
    const float* w2 = (const float*)d_in[4];
    float* out = (float*)d_out;

    char* ws = (char*)d_ws;
    const size_t MB = (size_t)1 << 20;
    int*   counts     = (int*)(ws);
    int*   cursors    = (int*)(ws + 32);
    int*   offsets    = (int*)(ws + 64);
    int*   tok_e      = (int*)(ws + 128);
    float* tok_w      = (float*)(ws + 128 + 65536);
    int*   row_token  = (int*)(ws + 128 + 2 * 65536);
    float* row_weight = (float*)(ws + 128 + 3 * 65536);
    unsigned short* xb   = (unsigned short*)(ws + 1 * MB);
    unsigned short* w1t  = (unsigned short*)(ws + 17 * MB);
    unsigned short* w3t  = (unsigned short*)(ws + 49 * MB);
    unsigned short* w2t  = (unsigned short*)(ws + 81 * MB);
    unsigned short* Hbuf = (unsigned short*)(ws + 113 * MB);

    hipMemsetAsync(ws, 0, 96, stream);
    hipMemsetAsync(d_out, 0, sizeof(float) * (size_t)out_size, stream);

    xconv_kernel<<<TOK * DIMC / 8 / 256, 256, 0, stream>>>(x, xb);
    wconv_kernel<DIMC, HIDC><<<NEXP * HIDC * (DIMC / 8) / 256, 256, 0, stream>>>(w1, w1t);
    wconv_kernel<DIMC, HIDC><<<NEXP * HIDC * (DIMC / 8) / 256, 256, 0, stream>>>(w3, w3t);
    wconv_kernel<HIDC, DIMC><<<NEXP * DIMC * (HIDC / 8) / 256, 256, 0, stream>>>(w2, w2t);

    gate_kernel<<<TOK / 8, 256, 0, stream>>>(x, gw, tok_e, tok_w, counts);
    scan_kernel<<<1, 32, 0, stream>>>(counts, offsets);
    scatter_kernel<<<TOK / 256, 256, 0, stream>>>(tok_e, tok_w, offsets, cursors,
                                                  row_token, row_weight);
    gemm1_kernel<<<dim3(HIDC / 128, TOK / 64, NEXP), 256, 0, stream>>>(
        xb, w1t, w3t, counts, offsets, row_token, Hbuf);
    gemm2_kernel<<<dim3(DIMC / 128, TOK / 64, NEXP), 256, 0, stream>>>(
        Hbuf, w2t, counts, offsets, row_token, row_weight, out);
}
